// NeuFilter_30631706755913
// MI455X (gfx1250) — compile-verified, matrix operand order checked by reading, not run on hardware
//
#include <hip/hip_runtime.h>
#include <math.h>

#define BD    8192
#define DD    256
#define FF    128
#define NROWS 500000
#define NBLK  (BD / 16)   // 512 fused blocks, one wave each

typedef __attribute__((ext_vector_type(16))) __bf16 v16bf;
typedef __attribute__((ext_vector_type(8)))  float  v8f;
typedef unsigned short u16;
typedef unsigned int   u32;

// ---- bf16 weight arena layout inside d_ws (after 512 f32 partial slots) ----
constexpr size_t OFF_TRAN_W0 = 0;        // 256x384
constexpr size_t OFF_TRAN_W1 = 98304;    // 256x256
constexpr size_t OFF_PRED_W0 = 163840;
constexpr size_t OFF_PRED_W1 = 229376;
constexpr size_t OFF_KIN_W0  = 294912;   // 256x256 (first 256 cols of 257)
constexpr size_t OFF_KIN_W1  = 360448;
constexpr size_t OFF_GRU_WIH = 425984;   // 768x256
constexpr size_t OFF_GRU_WHH = 622592;   // 768x256
constexpr size_t OFF_KOUT_W0 = 819200;
constexpr size_t OFF_KOUT_W1 = 884736;
constexpr size_t WTOT        = 950272;   // bf16 elements total

// ---------- conversion helpers (native casts -> v_cvt_pk_bf16_f32) ----------
__device__ __forceinline__ u16 f2bfu(float f) {
    union { __bf16 b; u16 u; } x; x.b = (__bf16)f; return x.u;
}
__device__ __forceinline__ float bf2f(u16 h) {
    union { u32 u; float f; } x; x.u = ((u32)h) << 16;
    return x.f;
}
__device__ __forceinline__ u32 pack2(float a, float b) {
    union { __bf16 h[2]; u32 u; } t;
    t.h[0] = (__bf16)a; t.h[1] = (__bf16)b; return t.u;
}

union FragU { v16bf v; u32 u[8]; };

// ---------- WMMA wrapper ----------
__device__ __forceinline__ v8f wmma_bf16(v16bf a, v16bf b, v8f c) {
    return __builtin_amdgcn_wmma_f32_16x16x32_bf16(false, a, false, b,
                                                   (short)0, c, false, false);
}

// ---------- A fragment from LDS bf16 row-major buffer (row stride 768, 16B aligned) ----
__device__ __forceinline__ v16bf load_frag_lds(const u16* buf, int kbase) {
    int lane = threadIdx.x & 31;
    int m = lane & 15;
    int s = (lane >> 4) * 8;
    const u32* p0 = (const u32*)(buf + m * 768 + kbase + s);
    const u32* p1 = (const u32*)(buf + m * 768 + kbase + s + 16);
    FragU r;
#pragma unroll
    for (int i = 0; i < 4; i++) { r.u[i] = p0[i]; r.u[4 + i] = p1[i]; }
    return r.v;
}

// ---------- B fragment: bf16 arena (fast) or f32 original (fallback) ----------
template <bool BF16W>
__device__ __forceinline__ v16bf load_wfrag(const float* __restrict__ Wf, int ldf,
                                            const u16* __restrict__ Wb, int ldb,
                                            int nbase, int kbase) {
    int lane = threadIdx.x & 31;
    int n = lane & 15;
    int s = (lane >> 4) * 8;
    FragU r;
    if (BF16W) {
        const u16* p = Wb + (size_t)(nbase + n) * ldb + kbase + s;
        const u32* q0 = (const u32*)p;
        const u32* q1 = (const u32*)(p + 16);
#pragma unroll
        for (int i = 0; i < 4; i++) { r.u[i] = q0[i]; r.u[4 + i] = q1[i]; }
    } else {
        const float* p0 = Wf + (size_t)(nbase + n) * ldf + kbase + s;
#pragma unroll
        for (int i = 0; i < 4; i++) {
            r.u[i]     = pack2(p0[2 * i],      p0[2 * i + 1]);
            r.u[4 + i] = pack2(p0[16 + 2 * i], p0[16 + 2 * i + 1]);
        }
    }
    return r.v;
}

// ---------- A fragment directly from gathered global f32 rows (h_table) ----------
__device__ __forceinline__ v16bf load_frag_hrows(const float* __restrict__ tab,
                                                 const int* rowids, int kbase) {
    int lane = threadIdx.x & 31;
    int m = lane & 15;
    int s = (lane >> 4) * 8;
    const float* p0 = tab + (size_t)rowids[m] * DD + kbase + s;
    FragU r;
#pragma unroll
    for (int i = 0; i < 4; i++) {
        r.u[i]     = pack2(p0[2 * i],      p0[2 * i + 1]);
        r.u[4 + i] = pack2(p0[16 + 2 * i], p0[16 + 2 * i + 1]);
    }
    return r.v;
}

// ---------- generic fused MLP layer: D = act(A @ W^T + b) ----------
template <int KT, bool RELU, bool BF16W>
__device__ __forceinline__ void mlp_layer(const u16* bufA, u16* bufD,
                                          const float* __restrict__ Wf, int ldf,
                                          const u16* __restrict__ Wb, int ldb,
                                          const float* __restrict__ bias, int Nout) {
    int lane = threadIdx.x & 31;
    int n = lane & 15, mo = (lane >> 4) * 8;
    v16bf a[KT];
#pragma unroll
    for (int i = 0; i < KT; i++) a[i] = load_frag_lds(bufA, i * 32);
    for (int nb = 0; nb < Nout; nb += 16) {
        v8f acc = {};
#pragma unroll
        for (int i = 0; i < KT; i++)
            acc = wmma_bf16(a[i], load_wfrag<BF16W>(Wf, ldf, Wb, ldb, nb, i * 32), acc);
        float bv = bias[nb + n];
#pragma unroll
        for (int r = 0; r < 8; r++) {
            float v = acc[r] + bv;
            if (RELU) v = fmaxf(v, 0.0f);
            bufD[(mo + r) * 768 + nb + n] = f2bfu(v);
        }
    }
}

// ================= kernel 0: f32 -> bf16 weight conversion into ws arena =============
__global__ void convert_w_kernel(const float* __restrict__ src, u16* __restrict__ dst,
                                 int rows, int cols, int ld_src) {
    int i = blockIdx.x * blockDim.x + threadIdx.x;
    int n = rows * cols;
    if (i < n) {
        int r = i / cols, c = i - r * cols;
        dst[i] = f2bfu(src[(size_t)r * ld_src + c]);
    }
}

// ================= kernel 1: bulk copy h_table -> h_new region of d_out ==============
__global__ void copy_table_kernel(const float* __restrict__ src, float* __restrict__ dst,
                                  long n4) {
    long i = (long)blockIdx.x * blockDim.x + threadIdx.x;
    if (i < n4) {
        float4 v = ((const float4*)src)[i];     // src is 16B aligned
        long b = i * 4;                          // dst only 4B aligned (offset B*D+1)
        dst[b] = v.x; dst[b + 1] = v.y; dst[b + 2] = v.z; dst[b + 3] = v.w;
    }
}

// ================= kernel 2: fused NeuralKF forward, one wave per 16-row tile =========
template <bool BF16W>
__global__ __launch_bounds__(32, 4) void neufilter_fused_kernel(
    const int* __restrict__ ids,
    const float* __restrict__ user_prior, const float* __restrict__ user_emb,
    const float* __restrict__ item_emb,   const float* __restrict__ feat,
    const float* __restrict__ h_table,
    const float* __restrict__ tran_W0, const float* __restrict__ tran_b0,
    const float* __restrict__ tran_W1, const float* __restrict__ tran_b1,
    const float* __restrict__ pred_W0, const float* __restrict__ pred_b0,
    const float* __restrict__ pred_W1, const float* __restrict__ pred_b1,
    const float* __restrict__ kin_W0,  const float* __restrict__ kin_b0,
    const float* __restrict__ kin_W1,  const float* __restrict__ kin_b1,
    const float* __restrict__ gru_Wih, const float* __restrict__ gru_Whh,
    const float* __restrict__ gru_bih, const float* __restrict__ gru_bhh,
    const float* __restrict__ kout_W0, const float* __restrict__ kout_b0,
    const float* __restrict__ kout_W1, const float* __restrict__ kout_b1,
    const u16* __restrict__ wb,
    float* __restrict__ out, float* __restrict__ h_new, float* __restrict__ ws_partial) {

    __shared__ u16 buf0[16 * 768];   // 24 KB ping
    __shared__ u16 buf1[16 * 768];   // 24 KB pong
    __shared__ u16 bufE[16 * 256];   // 8 KB  emb_post_pred (bf16)
    __shared__ float zres[16];
    __shared__ int   rowids[16];
    __shared__ float red[32];

    const int lane  = threadIdx.x & 31;
    const int n     = lane & 15;
    const int mo    = (lane >> 4) * 8;
    const int mbase = blockIdx.x * 16;

    // ---- stage 0: stage x0 = [user_emb | feat] into buf0 (bf16); gather ids; z_res ----
#pragma unroll 2
    for (int m = 0; m < 16; m++) {
        const float* ue = user_emb + (size_t)(mbase + m) * DD;
        const float* ft = feat     + (size_t)(mbase + m) * FF;
        u16* row = buf0 + m * 768;
        for (int c = lane; c < DD; c += 32) row[c] = f2bfu(ue[c]);
        for (int c = lane; c < FF; c += 32) row[DD + c] = f2bfu(ft[c]);
    }
    if (lane < 16) {
        rowids[lane] = ids[mbase + lane];
        const float4* ip = (const float4*)(item_emb   + (size_t)(mbase + lane) * DD);
        const float4* pp = (const float4*)(user_prior + (size_t)(mbase + lane) * DD);
        float s = 0.0f;
        for (int c = 0; c < DD / 4; c++) {
            float4 a = ip[c], b = pp[c];
            s += a.x * b.x + a.y * b.y + a.z * b.z + a.w * b.w;
        }
        zres[lane] = 1.0f - s;
    }
    __syncthreads();

    // ---- tran: h1 = relu(x0 @ W0^T + b0) ; emb = h1 @ W1^T + b1 ----
    mlp_layer<12, true,  BF16W>(buf0, buf1, tran_W0, 384, wb + OFF_TRAN_W0, 384, tran_b0, 256);
    __syncthreads();
    mlp_layer<8,  false, BF16W>(buf1, buf0, tran_W1, 256, wb + OFF_TRAN_W1, 256, tran_b1, 256);
    __syncthreads();
    // ---- pred: p1 = relu(emb @ W0^T + b0) ----
    mlp_layer<8,  true,  BF16W>(buf0, buf1, pred_W0, 256, wb + OFF_PRED_W0, 256, pred_b0, 256);
    __syncthreads();

    // ---- epp = p1 @ predW1^T + b1 ; keep epp (bufE) and emb_res = epp - prior (buf0) --
    {
        v16bf a[8];
#pragma unroll
        for (int i = 0; i < 8; i++) a[i] = load_frag_lds(buf1, i * 32);
        for (int nb = 0; nb < 256; nb += 16) {
            v8f acc = {};
#pragma unroll
            for (int i = 0; i < 8; i++)
                acc = wmma_bf16(a[i],
                    load_wfrag<BF16W>(pred_W1, 256, wb + OFF_PRED_W1, 256, nb, i * 32), acc);
            float bv = pred_b1[nb + n];
#pragma unroll
            for (int r = 0; r < 8; r++) {
                int m = mo + r;
                float v = acc[r] + bv;
                bufE[m * 256 + nb + n] = f2bfu(v);
                float res = v - user_prior[(size_t)(mbase + m) * DD + nb + n];
                buf0[m * 768 + nb + n] = f2bfu(res);
            }
        }
    }
    __syncthreads();

    // ---- k1 = relu([emb_res, z_res] @ kinW0^T + b0): K=256 GEMM + rank-1 z term ----
    {
        v16bf a[8];
#pragma unroll
        for (int i = 0; i < 8; i++) a[i] = load_frag_lds(buf0, i * 32);
        for (int nb = 0; nb < 256; nb += 16) {
            v8f acc = {};
#pragma unroll
            for (int i = 0; i < 8; i++)
                acc = wmma_bf16(a[i],
                    load_wfrag<BF16W>(kin_W0, 257, wb + OFF_KIN_W0, 256, nb, i * 32), acc);
            float bv = kin_b0[nb + n];
            float wz = kin_W0[(size_t)(nb + n) * 257 + 256];   // z_res column (f32)
#pragma unroll
            for (int r = 0; r < 8; r++) {
                int m = mo + r;
                float v = fmaxf(acc[r] + bv + zres[m] * wz, 0.0f);
                buf1[m * 768 + nb + n] = f2bfu(v);
            }
        }
    }
    __syncthreads();

    // ---- k_in = k1 @ kinW1^T + b1 ----
    mlp_layer<8, false, BF16W>(buf1, buf0, kin_W1, 256, wb + OFF_KIN_W1, 256, kin_b1, 256);
    __syncthreads();
    // ---- gi = k_in @ Wih^T + bih  (16x768) ----
    mlp_layer<8, false, BF16W>(buf0, buf1, gru_Wih, 256, wb + OFF_GRU_WIH, 256, gru_bih, 768);
    __syncthreads();

    // ---- gh = h @ Whh^T + bhh : A loaded straight from gathered h_table rows ----
    {
        v16bf a[8];
#pragma unroll
        for (int i = 0; i < 8; i++) a[i] = load_frag_hrows(h_table, rowids, i * 32);
        for (int nb = 0; nb < 768; nb += 16) {
            v8f acc = {};
#pragma unroll
            for (int i = 0; i < 8; i++)
                acc = wmma_bf16(a[i],
                    load_wfrag<BF16W>(gru_Whh, 256, wb + OFF_GRU_WHH, 256, nb, i * 32), acc);
            float bv = gru_bhh[nb + n];
#pragma unroll
            for (int r = 0; r < 8; r++)
                buf0[(mo + r) * 768 + nb + n] = f2bfu(acc[r] + bv);
        }
    }
    __syncthreads();

    // ---- GRU combine -> h_state (pairwise): scatter f32 to h_new, bf16 to buf0 ----
    {
        int m = lane & 15;
        int jb = (lane >> 4) * 128;
        size_t hrow = (size_t)rowids[m] * DD;
        const u16* gi = buf1 + m * 768;
        u16*       gh = buf0 + m * 768;
        for (int j = jb; j < jb + 128; j += 2) {
            u32 uir = *(const u32*)(gi + j);
            u32 uiz = *(const u32*)(gi + 256 + j);
            u32 uin = *(const u32*)(gi + 512 + j);
            u32 uhr = *(const u32*)(gh + j);
            u32 uhz = *(const u32*)(gh + 256 + j);
            u32 uhn = *(const u32*)(gh + 512 + j);
            float2 hh = *(const float2*)(h_table + hrow + j);
            float r0 = 1.0f / (1.0f + expf(-(bf2f((u16)uir) + bf2f((u16)uhr))));
            float r1 = 1.0f / (1.0f + expf(-(bf2f((u16)(uir >> 16)) + bf2f((u16)(uhr >> 16)))));
            float z0 = 1.0f / (1.0f + expf(-(bf2f((u16)uiz) + bf2f((u16)uhz))));
            float z1 = 1.0f / (1.0f + expf(-(bf2f((u16)(uiz >> 16)) + bf2f((u16)(uhz >> 16)))));
            float n0 = tanhf(bf2f((u16)uin) + r0 * bf2f((u16)uhn));
            float n1 = tanhf(bf2f((u16)(uin >> 16)) + r1 * bf2f((u16)(uhn >> 16)));
            float hs0 = (1.0f - z0) * n0 + z0 * hh.x;
            float hs1 = (1.0f - z1) * n1 + z1 * hh.y;
            *(float2*)(h_new + hrow + j) = make_float2(hs0, hs1);  // scatter
            *(u32*)(gh + j) = pack2(hs0, hs1);                     // stage for kout
        }
    }
    __syncthreads();

    // ---- kout: k2 = relu(hs @ W0^T + b0) ----
    mlp_layer<8, true, BF16W>(buf0, buf1, kout_W0, 256, wb + OFF_KOUT_W0, 256, kout_b0, 256);
    __syncthreads();

    // ---- K = k2 @ W1^T + b1 ; emb_post = prior + K*z ; regu partial ----
    float* ep32 = (float*)buf0;                  // 16x256 f32, buf0 is dead now
    float racc = 0.0f;
    {
        v16bf a[8];
#pragma unroll
        for (int i = 0; i < 8; i++) a[i] = load_frag_lds(buf1, i * 32);
        for (int nb = 0; nb < 256; nb += 16) {
            v8f acc = {};
#pragma unroll
            for (int i = 0; i < 8; i++)
                acc = wmma_bf16(a[i],
                    load_wfrag<BF16W>(kout_W1, 256, wb + OFF_KOUT_W1, 256, nb, i * 32), acc);
            float bv = kout_b1[nb + n];
#pragma unroll
            for (int r = 0; r < 8; r++) {
                int m = mo + r;
                float Kv = acc[r] + bv;
                float ep = user_prior[(size_t)(mbase + m) * DD + nb + n] + Kv * zres[m];
                float dd = bf2f(bufE[m * 256 + nb + n]) - ep;
                racc += dd * dd;
                ep32[m * 256 + nb + n] = ep;
            }
        }
    }
    __syncthreads();

    // ---- deterministic block regu partial ----
    red[lane] = racc;
    __syncthreads();
    if (lane == 0) {
        float s = 0.0f;
        for (int i = 0; i < 32; i++) s += red[i];
        ws_partial[blockIdx.x] = s;
    }

    // ---- row norms (deterministic 2-lane reduction) + normalized writeback (b128) ----
    {
        int m = lane & 15;
        int jb = (lane >> 4) * 128;
        const float4* ep4 = (const float4*)(ep32 + m * 256);
        float ps = 0.0f;
        for (int q = jb / 4; q < jb / 4 + 32; q++) {
            float4 v = ep4[q];
            ps += v.x * v.x + v.y * v.y + v.z * v.z + v.w * v.w;
        }
        red[lane] = ps;
        __syncthreads();
        float nrm = sqrtf(red[m] + red[m + 16]);
        float inv = 1.0f / fmaxf(nrm, 1e-12f);
        float4* o4 = (float4*)(out + (size_t)(mbase + m) * DD);
        for (int q = jb / 4; q < jb / 4 + 32; q++) {
            float4 v = ep4[q];
            o4[q] = make_float4(v.x * inv, v.y * inv, v.z * inv, v.w * inv);
        }
    }
}

// ================= kernel 3: deterministic regu reduction =================
__global__ void finalize_kernel(const float* __restrict__ ws, float* __restrict__ regu_out) {
    if (blockIdx.x == 0 && threadIdx.x == 0) {
        float s = 0.0f;
        for (int i = 0; i < NBLK; i++) s += ws[i];
        *regu_out = sqrtf(s);
    }
}

extern "C" void kernel_launch(void* const* d_in, const int* in_sizes, int n_in,
                              void* d_out, int out_size, void* d_ws, size_t ws_size,
                              hipStream_t stream) {
    (void)in_sizes; (void)n_in; (void)out_size;
    const int*   ids        = (const int*)  d_in[0];
    const float* user_prior = (const float*)d_in[1];
    const float* user_emb   = (const float*)d_in[2];
    const float* item_emb   = (const float*)d_in[3];
    const float* feat       = (const float*)d_in[4];
    const float* h_table    = (const float*)d_in[5];
    const float* tran_W0    = (const float*)d_in[6];
    const float* tran_b0    = (const float*)d_in[7];
    const float* tran_W1    = (const float*)d_in[8];
    const float* tran_b1    = (const float*)d_in[9];
    const float* pred_W0    = (const float*)d_in[10];
    const float* pred_b0    = (const float*)d_in[11];
    const float* pred_W1    = (const float*)d_in[12];
    const float* pred_b1    = (const float*)d_in[13];
    const float* kin_W0     = (const float*)d_in[14];
    const float* kin_b0     = (const float*)d_in[15];
    const float* kin_W1     = (const float*)d_in[16];
    const float* kin_b1     = (const float*)d_in[17];
    const float* gru_Wih    = (const float*)d_in[18];
    const float* gru_Whh    = (const float*)d_in[19];
    const float* gru_bih    = (const float*)d_in[20];
    const float* gru_bhh    = (const float*)d_in[21];
    const float* kout_W0    = (const float*)d_in[22];
    const float* kout_b0    = (const float*)d_in[23];
    const float* kout_W1    = (const float*)d_in[24];
    const float* kout_b1    = (const float*)d_in[25];

    float* out      = (float*)d_out;
    float* regu_out = out + (size_t)BD * DD;   // scalar at index B*D
    float* h_new    = regu_out + 1;            // table copy after the scalar
    float* ws       = (float*)d_ws;            // [0..511] f32 partials
    u16*   wb       = (u16*)((char*)d_ws + 2048);

    // 1) bulk copy table (stream order guarantees scatter rows overwrite later)
    long n4 = (long)NROWS * DD / 4;
    int cblocks = (int)((n4 + 255) / 256);
    copy_table_kernel<<<cblocks, 256, 0, stream>>>(h_table, h_new, n4);

    const size_t need = 2048 + (size_t)WTOT * 2;
    const bool usebf = (ws_size >= need);

    if (usebf) {
        // 0) per-call weight conversion into the bf16 arena
        auto cv = [&](const float* src, size_t off, int rows, int cols, int lds) {
            int nel = rows * cols;
            convert_w_kernel<<<(nel + 255) / 256, 256, 0, stream>>>(src, wb + off, rows, cols, lds);
        };
        cv(tran_W0, OFF_TRAN_W0, 256, 384, 384);
        cv(tran_W1, OFF_TRAN_W1, 256, 256, 256);
        cv(pred_W0, OFF_PRED_W0, 256, 256, 256);
        cv(pred_W1, OFF_PRED_W1, 256, 256, 256);
        cv(kin_W0,  OFF_KIN_W0,  256, 256, 257);   // drop z column (handled in f32)
        cv(kin_W1,  OFF_KIN_W1,  256, 256, 256);
        cv(gru_Wih, OFF_GRU_WIH, 768, 256, 256);
        cv(gru_Whh, OFF_GRU_WHH, 768, 256, 256);
        cv(kout_W0, OFF_KOUT_W0, 256, 256, 256);
        cv(kout_W1, OFF_KOUT_W1, 256, 256, 256);

        neufilter_fused_kernel<true><<<NBLK, 32, 0, stream>>>(
            ids, user_prior, user_emb, item_emb, feat, h_table,
            tran_W0, tran_b0, tran_W1, tran_b1,
            pred_W0, pred_b0, pred_W1, pred_b1,
            kin_W0, kin_b0, kin_W1, kin_b1,
            gru_Wih, gru_Whh, gru_bih, gru_bhh,
            kout_W0, kout_b0, kout_W1, kout_b1,
            wb, out, h_new, ws);
    } else {
        neufilter_fused_kernel<false><<<NBLK, 32, 0, stream>>>(
            ids, user_prior, user_emb, item_emb, feat, h_table,
            tran_W0, tran_b0, tran_W1, tran_b1,
            pred_W0, pred_b0, pred_W1, pred_b1,
            kin_W0, kin_b0, kin_W1, kin_b1,
            gru_Wih, gru_Whh, gru_bih, gru_bhh,
            kout_W0, kout_b0, kout_W1, kout_b1,
            wb, out, h_new, ws);
    }

    // 3) deterministic regu reduction
    finalize_kernel<<<1, 1, 0, stream>>>(ws, regu_out);
}
